// LSTM_73675868996126
// MI455X (gfx1250) — matrix operand order repaired and checked by reading, not compiled, not run on hardware
//
#include <hip/hip_runtime.h>
#include <hip/hip_bf16.h>

// ---------------------------------------------------------------------------
// 2-layer LSTM (T=4096, E=1024, H=2048, batch=1) + final linear/log-softmax.
// Strategy (MI455X):
//   1) convert weights/x to bf16 (recurrent weights: 96MB -> L2 resident)
//   2) precompute XG1[T,4H] = x @ W_ih1^T + (b_ih1+b_hh1) with WMMA bf16 GEMM
//   3) persistent kernel: 128 blocks own 16 h-lanes each (all 4 gates),
//      WMMA GEMV vs L2-resident bf16 W_hh1 / W_ih2 / W_hh2,
//      2 lock-free grid barriers per timestep, ping-pong h buffers
//   4) tiny epilogue kernel: W_out GEMV + log-softmax (fp32)
// ---------------------------------------------------------------------------

#define T_SEQ 4096
#define E_DIM 1024
#define H_DIM 2048
#define G_DIM 8192 /* 4*H */
#define NTAGS 128
#define NBLK 128   /* persistent blocks: 128*16 = 2048 = H */
#define PTHR 512   /* 16 waves per persistent block */

typedef __attribute__((ext_vector_type(16))) __bf16 v16bf;
typedef __attribute__((ext_vector_type(8)))  __bf16 v8bf;
typedef __attribute__((ext_vector_type(8)))  float  v8f;

__device__ __forceinline__ float sigmoidf_(float x) { return 1.0f / (1.0f + __expf(-x)); }

__device__ __forceinline__ v16bf make_afrag(const __bf16* __restrict__ row, int kb, int half) {
  // A 16x32 fragment, contiguous k order (B matches): lane (M) holds K = half*16 + e
  v8bf lo = *(const v8bf*)(row + kb + half * 16);
  v8bf hi = *(const v8bf*)(row + kb + half * 16 + 8);
  return __builtin_shufflevector(lo, hi, 0, 1, 2, 3, 4, 5, 6, 7, 8, 9, 10, 11, 12, 13, 14, 15);
}

// ---------------------------------------------------------------------------
// fp32 -> bf16 conversion (grid-stride)
// ---------------------------------------------------------------------------
__global__ void cvt_bf16_kernel(const float* __restrict__ s, __bf16* __restrict__ d, size_t n) {
  size_t i = (size_t)blockIdx.x * blockDim.x + threadIdx.x;
  size_t st = (size_t)gridDim.x * blockDim.x;
  for (; i < n; i += st) d[i] = (__bf16)s[i];
}

__global__ void vec_add_kernel(const float* __restrict__ a, const float* __restrict__ b,
                               float* __restrict__ o, int n) {
  int i = blockIdx.x * blockDim.x + threadIdx.x;
  if (i < n) o[i] = a[i] + b[i];
}

__global__ void init_state_kernel(__bf16* h1, __bf16* h2, float* c1, float* c2, unsigned* bar) {
  int i = blockIdx.x * blockDim.x + threadIdx.x;
  if (i < 2 * H_DIM) { h1[i] = (__bf16)0.0f; h2[i] = (__bf16)0.0f; }
  if (i < H_DIM) { c1[i] = 0.0f; c2[i] = 0.0f; }
  if (i < 2) bar[i] = 0u;
}

// ---------------------------------------------------------------------------
// XG1[T,4H] = x_bf16[T,E] @ W_ih1^T + bsum1   (WMMA bf16, fp32 accumulate)
// wave task = one 16-row M tile x four 16-col N tiles (A-frag reuse).
// tasks = 256 mtiles * 128 ngroups = 32768 waves = 4096 blocks * 8 waves
// ---------------------------------------------------------------------------
__global__ __launch_bounds__(256) void gemm_xgates_kernel(
    const __bf16* __restrict__ X, const __bf16* __restrict__ Wih1,
    const float* __restrict__ bsum1, float* __restrict__ XG1) {
  const int wtask = (blockIdx.x * blockDim.x + threadIdx.x) >> 5;
  const int lane = threadIdx.x & 31;
  const int half = lane >> 4, ln = lane & 15;
  const int ng = wtask & 127;         // 0..127  (64 output cols each)
  const int mt = wtask >> 7;          // 0..255  (16 output rows each)
  const int mrow = mt * 16;
  const int nbase0 = ng * 64;

  v8f acc[4] = {v8f{}, v8f{}, v8f{}, v8f{}};
  const __bf16* Xrow = X + (size_t)(mrow + ln) * E_DIM;

#pragma unroll 2
  for (int kb = 0; kb < E_DIM; kb += 32) {
    v16bf af = make_afrag(Xrow, kb, half);
#pragma unroll
    for (int j = 0; j < 4; ++j) {
      // B[k][n] = W_ih1^T[k][n] = W_ih1[n][k]; lane (N) holds K = half*16 + e
      const __bf16* Bp = Wih1 + (size_t)(nbase0 + j * 16 + ln) * E_DIM + kb + half * 16;
      v16bf bf = *(const v16bf*)Bp;
      acc[j] = __builtin_amdgcn_wmma_f32_16x16x32_bf16(false, af, false, bf, (short)0, acc[j],
                                                       false, false);
    }
  }
#pragma unroll
  for (int j = 0; j < 4; ++j) {
    const int n = nbase0 + j * 16 + ln;
    const float bias = bsum1[n];
#pragma unroll
    for (int r = 0; r < 8; ++r)  // D VGPR r <-> M = r + 8*half, N = ln
      XG1[(size_t)(mrow + 8 * half + r) * G_DIM + n] = acc[j][r] + bias;
  }
}

// ---------------------------------------------------------------------------
// WMMA GEMV tile: 16 weight rows x K range, vector broadcast across N columns.
// Returns fp32 partials; lanes with (lane&15)==0 carry column 0.
// ---------------------------------------------------------------------------
__device__ __forceinline__ v8f wmma_gemv(const __bf16* __restrict__ Wrow,
                                         const __bf16* __restrict__ vec, int kbeg, int kend,
                                         int half) {
  v8f a0 = {}, a1 = {};
  for (int kb = kbeg; kb < kend; kb += 64) {
    __builtin_prefetch(Wrow + kb + 256, 0, 1);  // global_prefetch_b8 (L2-resident stream)
    {
      v16bf bf = *(const v16bf*)(vec + kb + half * 16);
      v16bf af = make_afrag(Wrow, kb, half);
      a0 = __builtin_amdgcn_wmma_f32_16x16x32_bf16(false, af, false, bf, (short)0, a0, false,
                                                   false);
    }
    {
      v16bf bf = *(const v16bf*)(vec + kb + 32 + half * 16);
      v16bf af = make_afrag(Wrow, kb + 32, half);
      a1 = __builtin_amdgcn_wmma_f32_16x16x32_bf16(false, af, false, bf, (short)0, a1, false,
                                                   false);
    }
  }
  return a0 + a1;
}

// ---------------------------------------------------------------------------
// lock-free grid barrier (arrive counter + generation), s_sleep in spin
// ---------------------------------------------------------------------------
__device__ __forceinline__ void grid_sync(unsigned* cnt, unsigned* gen, unsigned nb) {
  __syncthreads();
  if (threadIdx.x == 0) {
    unsigned g = __hip_atomic_load(gen, __ATOMIC_ACQUIRE, __HIP_MEMORY_SCOPE_AGENT);
    unsigned prev = __hip_atomic_fetch_add(cnt, 1u, __ATOMIC_ACQ_REL, __HIP_MEMORY_SCOPE_AGENT);
    if (prev == nb - 1u) {
      __hip_atomic_store(cnt, 0u, __ATOMIC_RELAXED, __HIP_MEMORY_SCOPE_AGENT);
      __hip_atomic_fetch_add(gen, 1u, __ATOMIC_ACQ_REL, __HIP_MEMORY_SCOPE_AGENT);
    } else {
      while (__hip_atomic_load(gen, __ATOMIC_ACQUIRE, __HIP_MEMORY_SCOPE_AGENT) == g)
        __builtin_amdgcn_s_sleep(2);
    }
  }
  __syncthreads();
}

// ---------------------------------------------------------------------------
// Persistent recurrence. Block b owns h-indices [16b, 16b+16) and computes all
// 4 gates for them. 16 waves = 4 gate-tiles x 4 K-splits; LDS reduce.
// 2 grid barriers per timestep; h1/h2 ping-pong buffers.
// ---------------------------------------------------------------------------
__global__ __launch_bounds__(PTHR, 1) void lstm_persistent_kernel(
    const float* __restrict__ XG1, const __bf16* __restrict__ Whh1,
    const __bf16* __restrict__ Wih2, const __bf16* __restrict__ Whh2,
    const float* __restrict__ bsum2, __bf16* __restrict__ h1buf, __bf16* __restrict__ h2buf,
    float* __restrict__ c1, float* __restrict__ c2, unsigned* __restrict__ bar) {
  const int b = blockIdx.x;
  const int tid = threadIdx.x;
  const int wave = tid >> 5, lane = tid & 31;
  const int half = lane >> 4, ln = lane & 15;
  const int gt = wave >> 2;  // gate tile: 0=i 1=f 2=g 3=o
  const int ks = wave & 3;   // K split
  const int hbase = b * 16;

  __shared__ float red[16 * 16];

  const __bf16* W1row = Whh1 + (size_t)(gt * H_DIM + hbase + ln) * H_DIM;
  const __bf16* W2row =
      (ks < 2 ? Wih2 : Whh2) + (size_t)(gt * H_DIM + hbase + ln) * H_DIM;

  for (int t = 0; t < T_SEQ; ++t) {
    const int cur = t & 1, prev = cur ^ 1;
    const __bf16* h1prev = h1buf + prev * H_DIM;
    const __bf16* h1cur = h1buf + cur * H_DIM;
    const __bf16* h2prev = h2buf + prev * H_DIM;

    // ---- layer 1: gates1 = W_hh1 @ h1_prev (+XG1 which holds x-part+bias) ----
    {
      v8f acc = wmma_gemv(W1row, h1prev, ks * 512, ks * 512 + 512, half);
      if (ln == 0) {
#pragma unroll
        for (int r = 0; r < 8; ++r) red[wave * 16 + 8 * half + r] = acc[r];
      }
    }
    __syncthreads();
    if (tid < 16) {
      const int r = tid;
      float s[4];
#pragma unroll
      for (int g = 0; g < 4; ++g) {
        float v = 0.0f;
#pragma unroll
        for (int k2 = 0; k2 < 4; ++k2) v += red[(g * 4 + k2) * 16 + r];
        s[g] = v + XG1[(size_t)t * G_DIM + g * H_DIM + hbase + r];
      }
      float cv = sigmoidf_(s[1]) * c1[hbase + r] + sigmoidf_(s[0]) * tanhf(s[2]);
      c1[hbase + r] = cv;
      h1buf[cur * H_DIM + hbase + r] = (__bf16)(sigmoidf_(s[3]) * tanhf(cv));
    }
    grid_sync(&bar[0], &bar[1], NBLK);  // h1(t) complete & visible

    // ---- layer 2: gates2 = W_ih2 @ h1(t) + W_hh2 @ h2(t-1) + bsum2 ----
    {
      const __bf16* vec = (ks < 2) ? h1cur : h2prev;
      const int k0 = (ks & 1) * 1024;
      v8f acc = wmma_gemv(W2row, vec, k0, k0 + 1024, half);
      if (ln == 0) {
#pragma unroll
        for (int r = 0; r < 8; ++r) red[wave * 16 + 8 * half + r] = acc[r];
      }
    }
    __syncthreads();
    if (tid < 16) {
      const int r = tid;
      float s[4];
#pragma unroll
      for (int g = 0; g < 4; ++g) {
        float v = 0.0f;
#pragma unroll
        for (int k2 = 0; k2 < 4; ++k2) v += red[(g * 4 + k2) * 16 + r];
        s[g] = v + bsum2[g * H_DIM + hbase + r];
      }
      float cv = sigmoidf_(s[1]) * c2[hbase + r] + sigmoidf_(s[0]) * tanhf(s[2]);
      c2[hbase + r] = cv;
      h2buf[cur * H_DIM + hbase + r] = (__bf16)(sigmoidf_(s[3]) * tanhf(cv));
    }
    grid_sync(&bar[0], &bar[1], NBLK);  // h2(t) complete & visible
  }
}

// ---------------------------------------------------------------------------
// out[128] = log_softmax(h2 @ W_out^T + b_out)
// ---------------------------------------------------------------------------
__global__ void tag_logsoftmax_kernel(const __bf16* __restrict__ h2,
                                      const float* __restrict__ Wout,
                                      const float* __restrict__ bout, float* __restrict__ out) {
  const int j = threadIdx.x;  // 128 threads
  const float* wr = Wout + (size_t)j * H_DIM;
  float acc = bout[j];
  for (int k = 0; k < H_DIM; ++k) acc += (float)h2[k] * wr[k];

  __shared__ float sm[NTAGS];
  sm[j] = acc;
  __syncthreads();
  for (int s = NTAGS / 2; s > 0; s >>= 1) {
    if (j < s) sm[j] = fmaxf(sm[j], sm[j + s]);
    __syncthreads();
  }
  const float mx = sm[0];
  __syncthreads();
  sm[j] = __expf(acc - mx);
  __syncthreads();
  for (int s = NTAGS / 2; s > 0; s >>= 1) {
    if (j < s) sm[j] += sm[j + s];
    __syncthreads();
  }
  out[j] = acc - (mx + __logf(sm[0]));
}

// ---------------------------------------------------------------------------
extern "C" void kernel_launch(void* const* d_in, const int* in_sizes, int n_in, void* d_out,
                              int out_size, void* d_ws, size_t ws_size, hipStream_t stream) {
  const float* x = (const float*)d_in[0];
  const float* Wih1 = (const float*)d_in[1];
  const float* Whh1 = (const float*)d_in[2];
  const float* bih1 = (const float*)d_in[3];
  const float* bhh1 = (const float*)d_in[4];
  const float* Wih2 = (const float*)d_in[5];
  const float* Whh2 = (const float*)d_in[6];
  const float* bih2 = (const float*)d_in[7];
  const float* bhh2 = (const float*)d_in[8];
  const float* Wout = (const float*)d_in[9];
  const float* bout = (const float*)d_in[10];

  size_t off = 0;
  auto take = [&](size_t bytes) -> char* {
    char* r = (char*)d_ws + off;
    off += (bytes + 255) & ~(size_t)255;
    return r;
  };
  __bf16* xbf = (__bf16*)take((size_t)T_SEQ * E_DIM * 2);
  __bf16* wih1b = (__bf16*)take((size_t)G_DIM * E_DIM * 2);
  __bf16* whh1b = (__bf16*)take((size_t)G_DIM * H_DIM * 2);
  __bf16* wih2b = (__bf16*)take((size_t)G_DIM * H_DIM * 2);
  __bf16* whh2b = (__bf16*)take((size_t)G_DIM * H_DIM * 2);
  float* bsum1 = (float*)take((size_t)G_DIM * 4);
  float* bsum2 = (float*)take((size_t)G_DIM * 4);
  __bf16* h1buf = (__bf16*)take((size_t)2 * H_DIM * 2);
  __bf16* h2buf = (__bf16*)take((size_t)2 * H_DIM * 2);
  float* c1 = (float*)take((size_t)H_DIM * 4);
  float* c2 = (float*)take((size_t)H_DIM * 4);
  unsigned* bar = (unsigned*)take(256);
  float* XG1 = (float*)take((size_t)T_SEQ * G_DIM * 4);

  // 1) conversions + bias folds + state init
  cvt_bf16_kernel<<<2048, 256, 0, stream>>>(x, xbf, (size_t)T_SEQ * E_DIM);
  cvt_bf16_kernel<<<4096, 256, 0, stream>>>(Wih1, wih1b, (size_t)G_DIM * E_DIM);
  cvt_bf16_kernel<<<4096, 256, 0, stream>>>(Whh1, whh1b, (size_t)G_DIM * H_DIM);
  cvt_bf16_kernel<<<4096, 256, 0, stream>>>(Wih2, wih2b, (size_t)G_DIM * H_DIM);
  cvt_bf16_kernel<<<4096, 256, 0, stream>>>(Whh2, whh2b, (size_t)G_DIM * H_DIM);
  vec_add_kernel<<<G_DIM / 256, 256, 0, stream>>>(bih1, bhh1, bsum1, G_DIM);
  vec_add_kernel<<<G_DIM / 256, 256, 0, stream>>>(bih2, bhh2, bsum2, G_DIM);
  init_state_kernel<<<16, 256, 0, stream>>>(h1buf, h2buf, c1, c2, bar);

  // 2) XG1 = x @ W_ih1^T + bsum1  (WMMA GEMM, 32768 wave-tasks)
  gemm_xgates_kernel<<<4096, 256, 0, stream>>>(xbf, wih1b, bsum1, XG1);

  // 3) persistent sequential recurrence (L2-resident bf16 weights)
  lstm_persistent_kernel<<<NBLK, PTHR, 0, stream>>>(XG1, whh1b, wih2b, whh2b, bsum2, h1buf,
                                                    h2buf, c1, c2, bar);

  // 4) final linear + log-softmax on last h2 (parity of step T-1)
  const __bf16* h2fin = h2buf + (size_t)((T_SEQ - 1) & 1) * H_DIM;
  tag_logsoftmax_kernel<<<1, NTAGS, 0, stream>>>(h2fin, Wout, bout, (float*)d_out);
}
